// ScanGRU_12249246728590
// MI455X (gfx1250) — matrix operand order repaired and checked by reading, not compiled, not run on hardware
//
#include <hip/hip_runtime.h>
#include <hip/hip_bf16.h>
#include <stdint.h>

// Problem constants (from reference): S=2048, B=64, I=H=512, L=2
#define S_TOTAL 2048
#define BATCH   64
#define HID     512
#define G3      1536          // 3*HID gate columns
#define NLAYERS 2
#define CHUNK_S 256           // timesteps of x_lin precomputed per chunk
#define NWG     8             // scan workgroups = HID / JW
#define JW      64            // hidden columns per scan workgroup

// LDS row strides in dwords: multiples of 4 (16B alignment for b128 loads),
// chosen so wave-wide fragment reads hit every bank exactly twice (no serialization).
#define STRIDE_CHUNK 20       // 32 bf16 payload (16 dwords) + pad
#define STRIDE_H     260      // 512 bf16 payload (256 dwords) + pad

typedef __attribute__((ext_vector_type(16))) __bf16 v16bf;
typedef __attribute__((ext_vector_type(8)))  float  v8f;

union Frag { v16bf v; unsigned u[8]; };

// ---------- scalar helpers ----------
__device__ __forceinline__ unsigned short f2bf(float f) {
  unsigned x = __float_as_uint(f);
  x += 0x7FFFu + ((x >> 16) & 1u);          // round-to-nearest-even
  return (unsigned short)(x >> 16);
}
__device__ __forceinline__ unsigned pack2bf(float lo, float hi) {
  return (unsigned)f2bf(lo) | ((unsigned)f2bf(hi) << 16);
}
__device__ __forceinline__ float sigmoidf_(float x) {
  return 1.0f / (1.0f + __expf(-x));
}
__device__ __forceinline__ float tanhf_(float x) {
  float e = __expf(2.0f * x);
  return (e - 1.0f) / (e + 1.0f);
}

// ---------- WMMA fragment loads (wave32, 16x16x32 bf16) ----------
// A 16x32 (ISA 7.12.2 dense table): row = lane&15 for all lanes;
//   lanes0-15: v0..3 -> K 0..7, v4..7 -> K 16..23 ; lanes16-31: +8.
//   Two contiguous 4-dword runs at dword offsets {hi?4:0} and {+8} -> 2x ds_load_b128.
__device__ __forceinline__ void load_frag_a(Frag& f, const unsigned* base,
                                            int row, int strideU, int hi) {
  const uint4* p = (const uint4*)(base + (size_t)row * strideU + (hi ? 4 : 0));
  uint4 q0 = p[0];          // K run 0
  uint4 q1 = p[2];          // K run 1 (+8 dwords)
  f.u[0] = q0.x; f.u[1] = q0.y; f.u[2] = q0.z; f.u[3] = q0.w;
  f.u[4] = q1.x; f.u[5] = q1.y; f.u[6] = q1.z; f.u[7] = q1.w;
}
// B 32x16: col = lane&15; same K order as A:
//   lanes0-15: v0..3 -> K 0..7, v4..7 -> K 16..23 ; lanes16-31: +8.
//   Two contiguous 4-dword runs at dword offsets {hi?4:0} and {+8} -> 2x ds_load_b128.
__device__ __forceinline__ void load_frag_b(Frag& f, const unsigned* base,
                                            int row, int strideU, int hi) {
  const uint4* p = (const uint4*)(base + (size_t)row * strideU + (hi ? 4 : 0));
  uint4 q0 = p[0];
  uint4 q1 = p[2];
  f.u[0] = q0.x; f.u[1] = q0.y; f.u[2] = q0.z; f.u[3] = q0.w;
  f.u[4] = q1.x; f.u[5] = q1.y; f.u[6] = q1.z; f.u[7] = q1.w;
}
__device__ __forceinline__ v8f wmma_bf(const Frag& a, const Frag& b, v8f c) {
  return __builtin_amdgcn_wmma_f32_16x16x32_bf16(
      false, a.v, false, b.v, (short)0, c, false, false);
}

// ---------- elementwise kernels ----------
__global__ void k_cvt_bf16(const float* __restrict__ src,
                           unsigned short* __restrict__ dst, long n) {
  long i = (long)blockIdx.x * blockDim.x + threadIdx.x;
  long stride = (long)gridDim.x * blockDim.x;
  for (; i < n; i += stride) dst[i] = f2bf(src[i]);
}
__global__ void k_copy_f32(const float* __restrict__ src,
                           float* __restrict__ dst, int n) {
  int i = blockIdx.x * blockDim.x + threadIdx.x;
  if (i < n) dst[i] = src[i];
}

// ---------- x_lin GEMM: [rows x HID](bf16) x [G3 x HID]^T(bf16) + bias ----------
// Grid: (CHUNK_S*BATCH/64, HID/JW), 256 threads = 8 waves.
// Wave (jt = w&3, mg = w>>2) owns m-tiles {mg*32, mg*32+16} x j-subtile jt,
// for all 3 gates -> 6 accumulators.
__global__ __launch_bounds__(256)
void k_gemm_xlin(const unsigned short* __restrict__ seq_bf,  // [S*B][HID]
                 const unsigned short* __restrict__ wih_b,   // [G3][HID]
                 const float* __restrict__ bih,               // [G3]
                 float* __restrict__ xlin,                    // [chunk rows][G3]
                 int rowBase) {
  __shared__ __align__(16) unsigned a_lds[64  * STRIDE_CHUNK];
  __shared__ __align__(16) unsigned b_lds[192 * STRIDE_CHUNK];

  const int tid = threadIdx.x;
  const int lane = tid & 31, wave = tid >> 5;
  const int lo = lane & 15, hi = lane >> 4;
  const int jt = wave & 3, mg = wave >> 2;
  const int m0g = rowBase + blockIdx.x * 64;   // global flat (s*B+b) row
  const int m0l = blockIdx.x * 64;             // chunk-local row
  const int j0 = blockIdx.y * JW;

  const unsigned* seq_u = (const unsigned*)seq_bf;   // 256 dwords / row
  const unsigned* wih_u = (const unsigned*)wih_b;

  v8f z = {0.f,0.f,0.f,0.f,0.f,0.f,0.f,0.f};
  v8f acc[2][3];
#pragma unroll
  for (int i = 0; i < 2; ++i)
#pragma unroll
    for (int g = 0; g < 3; ++g) acc[i][g] = z;

  for (int kc = 0; kc < HID; kc += 32) {
    // stage A chunk: 64x32 bf16, coalesced dword loads
    for (int i = tid; i < 64 * 16; i += 256) {
      int rl = i >> 4, c = i & 15;
      a_lds[rl * STRIDE_CHUNK + c] = seq_u[(size_t)(m0g + rl) * 256 + (kc >> 1) + c];
    }
    // stage B chunk: 192 gate-rows x 32 bf16
    for (int i = tid; i < 192 * 16; i += 256) {
      int rl = i >> 4, c = i & 15;
      int grow = (rl >> 6) * HID + j0 + (rl & 63);
      b_lds[rl * STRIDE_CHUNK + c] = wih_u[(size_t)grow * 256 + (kc >> 1) + c];
    }
    __syncthreads();
    Frag a0, a1, b0, b1, b2;
    load_frag_a(a0, a_lds, mg * 32 + lo,       STRIDE_CHUNK, hi);
    load_frag_a(a1, a_lds, mg * 32 + 16 + lo,  STRIDE_CHUNK, hi);
    load_frag_b(b0, b_lds, 0 * 64 + jt * 16 + lo, STRIDE_CHUNK, hi);
    load_frag_b(b1, b_lds, 1 * 64 + jt * 16 + lo, STRIDE_CHUNK, hi);
    load_frag_b(b2, b_lds, 2 * 64 + jt * 16 + lo, STRIDE_CHUNK, hi);
    acc[0][0] = wmma_bf(a0, b0, acc[0][0]);
    acc[0][1] = wmma_bf(a0, b1, acc[0][1]);
    acc[0][2] = wmma_bf(a0, b2, acc[0][2]);
    acc[1][0] = wmma_bf(a1, b0, acc[1][0]);
    acc[1][1] = wmma_bf(a1, b1, acc[1][1]);
    acc[1][2] = wmma_bf(a1, b2, acc[1][2]);
    __syncthreads();
  }

  const int j_glob = j0 + jt * 16 + lo;
#pragma unroll
  for (int mt = 0; mt < 2; ++mt) {
#pragma unroll
    for (int g = 0; g < 3; ++g) {
      int col = g * HID + j_glob;
      float bias = bih[col];
#pragma unroll
      for (int r = 0; r < 8; ++r) {
        int m = mg * 32 + mt * 16 + hi * 8 + r;
        xlin[(size_t)(m0l + m) * G3 + col] = acc[mt][g][r] + bias;
      }
    }
  }
}

// ---------- persistent recurrent scan ----------
// 8 workgroups, each owns 64 hidden columns (x3 gates). Per step:
//   A) full h (fp32, global ping-pong) -> bf16 in LDS
//   B) h @ whh^T via WMMA, whh chunk staged to LDS per k-step
//   C) gates: each lane already holds matching r/z/n pre-activations
//   D) device-scope barrier (monotonic atomic counter)
__global__ __launch_bounds__(256)
void k_scan(const unsigned short* __restrict__ whh_b,  // [G3][HID] bf16
            const float* __restrict__ xlin,            // [CHUNK_S][BATCH][G3]
            const float* __restrict__ bhh,             // [G3]
            float* __restrict__ hbuf,                  // [2][BATCH][HID] ping-pong
            unsigned short* __restrict__ out_bf,       // layer0: bf16 seq for next layer
            float* __restrict__ out_f32,               // layer1: d_out output
            float* __restrict__ hid_out,               // final hidden [BATCH][HID]
            unsigned* __restrict__ bar,
            int sBeg, int sEnd) {
  __shared__ __align__(16) unsigned h_lds[BATCH * STRIDE_H];   // ~66.6 KB
  __shared__ __align__(16) unsigned w_lds[192 * STRIDE_CHUNK]; // ~15.4 KB

  const int tid = threadIdx.x;
  const int lane = tid & 31, wave = tid >> 5;
  const int lo = lane & 15, hi = lane >> 4;
  const int jt = wave & 3, mg = wave >> 2;
  const int j0 = blockIdx.x * JW;
  const int j_glob = j0 + jt * 16 + lo;

  const float br = bhh[j_glob];
  const float bz = bhh[HID + j_glob];
  const float bn = bhh[2 * HID + j_glob];

  const unsigned* whh_u = (const unsigned*)whh_b;   // 256 dwords / row
  const v8f zero8 = {0.f,0.f,0.f,0.f,0.f,0.f,0.f,0.f};

  for (int s = sBeg; s < sEnd; ++s) {
    const float* hin  = hbuf + (size_t)(s & 1) * (BATCH * HID);
    float*       hout = hbuf + (size_t)((s + 1) & 1) * (BATCH * HID);

    // ---- A: stage full h as bf16 into LDS ----
    const float2* h2 = (const float2*)hin;
    for (int i = tid; i < BATCH * 256; i += 256) {
      int row = i >> 8, c = i & 255;
      float2 p = h2[row * 256 + c];
      h_lds[row * STRIDE_H + c] = pack2bf(p.x, p.y);
    }

    v8f acc[2][3];
#pragma unroll
    for (int i = 0; i < 2; ++i)
#pragma unroll
      for (int g = 0; g < 3; ++g) acc[i][g] = zero8;

    // ---- B: K=512 reduction in 32-wide chunks ----
    for (int kc = 0; kc < HID; kc += 32) {
      for (int i = tid; i < 192 * 16; i += 256) {
        int rl = i >> 4, c = i & 15;
        int grow = (rl >> 6) * HID + j0 + (rl & 63);
        w_lds[rl * STRIDE_CHUNK + c] = whh_u[(size_t)grow * 256 + (kc >> 1) + c];
      }
      __syncthreads();
      Frag a0, a1, b0, b1, b2;
      const unsigned* hbase = h_lds + (kc >> 1);   // +16 dwords per k-chunk: 16B aligned
      load_frag_a(a0, hbase, mg * 32 + lo,      STRIDE_H, hi);
      load_frag_a(a1, hbase, mg * 32 + 16 + lo, STRIDE_H, hi);
      load_frag_b(b0, w_lds, 0 * 64 + jt * 16 + lo, STRIDE_CHUNK, hi);
      load_frag_b(b1, w_lds, 1 * 64 + jt * 16 + lo, STRIDE_CHUNK, hi);
      load_frag_b(b2, w_lds, 2 * 64 + jt * 16 + lo, STRIDE_CHUNK, hi);
      acc[0][0] = wmma_bf(a0, b0, acc[0][0]);
      acc[0][1] = wmma_bf(a0, b1, acc[0][1]);
      acc[0][2] = wmma_bf(a0, b2, acc[0][2]);
      acc[1][0] = wmma_bf(a1, b0, acc[1][0]);
      acc[1][1] = wmma_bf(a1, b1, acc[1][1]);
      acc[1][2] = wmma_bf(a1, b2, acc[1][2]);
      __syncthreads();
    }

    // ---- C: gates (each lane holds r/z/n for same (b, j_glob)) ----
    const float* xb = xlin + (size_t)(s - sBeg) * (BATCH * G3);
    if (s + 1 < sEnd) {   // L2 prefetch hint for next step's x_lin row
      const float* xn = xb + BATCH * G3;
      __builtin_prefetch(xn + (size_t)(mg * 32 + hi * 8) * G3 + j_glob, 0, 1);
    }
#pragma unroll
    for (int mt = 0; mt < 2; ++mt) {
#pragma unroll
      for (int r = 0; r < 8; ++r) {
        int b = mg * 32 + mt * 16 + hi * 8 + r;
        const float* xr = xb + (size_t)b * G3;
        float hr = acc[mt][0][r] + br;
        float hz = acc[mt][1][r] + bz;
        float hn = acc[mt][2][r] + bn;
        float rg = sigmoidf_(xr[j_glob] + hr);
        float zg = sigmoidf_(xr[HID + j_glob] + hz);
        float ng = tanhf_(xr[2 * HID + j_glob] + rg * hn);
        float hp = hin[b * HID + j_glob];
        float hnv = (1.0f - zg) * ng + zg * hp;
        hout[b * HID + j_glob] = hnv;
        if (out_bf)
          out_bf[((size_t)s * BATCH + b) * HID + j_glob] = f2bf(hnv);
        if (out_f32)
          out_f32[((size_t)s * BATCH + b) * HID + j_glob] = hnv;
        if (s == S_TOTAL - 1) hid_out[b * HID + j_glob] = hnv;
      }
    }

    // ---- D: device-wide barrier (monotonic counter, no reset) ----
    __syncthreads();
    if (tid == 0) {
      __hip_atomic_fetch_add(bar, 1u, __ATOMIC_RELEASE, __HIP_MEMORY_SCOPE_AGENT);
      unsigned target = (unsigned)(s + 1) * NWG;
      while (__hip_atomic_load(bar, __ATOMIC_ACQUIRE, __HIP_MEMORY_SCOPE_AGENT) < target)
        __builtin_amdgcn_s_sleep(2);
    }
    __syncthreads();
  }
}

// ---------- host orchestration ----------
extern "C" void kernel_launch(void* const* d_in, const int* in_sizes, int n_in,
                              void* d_out, int out_size, void* d_ws, size_t ws_size,
                              hipStream_t stream) {
  (void)in_sizes; (void)n_in; (void)out_size; (void)ws_size;
  const float* x    = (const float*)d_in[0];   // [S,B,I]
  const float* hx   = (const float*)d_in[1];   // [L,B,H]
  const float* w_ih = (const float*)d_in[2];   // [L,3H,I]
  const float* w_hh = (const float*)d_in[3];   // [L,3H,H]
  const float* b_ih = (const float*)d_in[4];   // [L,3H]
  const float* b_hh = (const float*)d_in[5];   // [L,3H]
  float* out = (float*)d_out;                               // [S,B,H]
  float* hid = out + (size_t)S_TOTAL * BATCH * HID;         // [L,B,H]

  // workspace layout (~241 MB total)
  char* p = (char*)d_ws;
  unsigned short* wih_b  = (unsigned short*)p; p += (size_t)NLAYERS * G3 * HID * 2;
  unsigned short* whh_b  = (unsigned short*)p; p += (size_t)NLAYERS * G3 * HID * 2;
  unsigned short* seq_bf = (unsigned short*)p; p += (size_t)S_TOTAL * BATCH * HID * 2;
  float*          xlin   = (float*)p;          p += (size_t)CHUNK_S * BATCH * G3 * 4;
  float*          hbuf   = (float*)p;          p += (size_t)2 * BATCH * HID * 4;
  unsigned*       bar    = (unsigned*)p;       p += 256;

  hipMemsetAsync(bar, 0, 256, stream);

  long nw = (long)NLAYERS * G3 * HID;
  k_cvt_bf16<<<1024, 256, 0, stream>>>(w_ih, wih_b, nw);
  k_cvt_bf16<<<1024, 256, 0, stream>>>(w_hh, whh_b, nw);
  k_cvt_bf16<<<4096, 256, 0, stream>>>(x, seq_bf, (long)S_TOTAL * BATCH * HID);

  for (int l = 0; l < NLAYERS; ++l) {
    k_copy_f32<<<(BATCH * HID + 255) / 256, 256, 0, stream>>>(
        hx + (size_t)l * BATCH * HID, hbuf, BATCH * HID);
    for (int c = 0; c < S_TOTAL / CHUNK_S; ++c) {
      int sBeg = c * CHUNK_S;
      k_gemm_xlin<<<dim3(CHUNK_S * BATCH / 64, HID / JW), 256, 0, stream>>>(
          seq_bf, wih_b + (size_t)l * G3 * HID, b_ih + (size_t)l * G3,
          xlin, sBeg * BATCH);
      k_scan<<<NWG, 256, 0, stream>>>(
          whh_b + (size_t)l * G3 * HID, xlin, b_hh + (size_t)l * G3, hbuf,
          (l == 0) ? seq_bf : (unsigned short*)nullptr,
          (l == 1) ? out : (float*)nullptr,
          hid + (size_t)l * BATCH * HID,
          bar + l * 32, sBeg, sBeg + CHUNK_S);
    }
  }
}
